// MultiHeadAttention_79285096284234
// MI455X (gfx1250) — matrix operand order repaired and checked by reading, not compiled, not run on hardware
//
#include <hip/hip_runtime.h>

#define DM 256   // d_model
#define GD 1024  // 4*d_model (gates)
#define SQ 1024  // sequence length
#define BB 4     // batch

typedef __attribute__((ext_vector_type(16))) _Float16 v16h;
typedef __attribute__((ext_vector_type(8)))  _Float16 v8h;
typedef __attribute__((ext_vector_type(8)))  float    v8f;
typedef unsigned int u32x4 __attribute__((ext_vector_type(4)));
typedef int          i32x8 __attribute__((ext_vector_type(8)));
typedef int          i32x4 __attribute__((ext_vector_type(4)));

union HFrag { v16h v; v8h h[2]; };

// A-fragment (16x32 f16): this lane covers row m (caller passes row ptr).
__device__ __forceinline__ v16h load_a_frag(const _Float16* row, int k0, int grp) {
  HFrag f;
  f.h[0] = *(const v8h*)(row + k0 + 8 * grp);
  f.h[1] = *(const v8h*)(row + k0 + 16 + 8 * grp);
  return f.v;
}
// B-fragment (32x16 f16): lane covers column n (caller passes K-major column ptr).
__device__ __forceinline__ v16h load_b_frag(const _Float16* col, int k0, int grp) {
  HFrag f;
  f.h[0] = *(const v8h*)(col + k0 + 8 * grp);
  f.h[1] = *(const v8h*)(col + k0 + 16 + 8 * grp);
  return f.v;
}
__device__ __forceinline__ v8f wmma16(v16h a, v16h b, v8f c) {
  return __builtin_amdgcn_wmma_f32_16x16x32_f16(false, a, false, b, (short)0, c,
                                                false, false);
}
__device__ __forceinline__ float sigf(float x) { return 1.0f / (1.0f + __expf(-x)); }

// ---- Tensor Data Mover: async 2-D tile load (4 rows x 1024 f32) into LDS ----
// D# per CDNA5 ISA 08_async_tensor 8.3/8.4. 2-D tensor: groups 2/3 zero.
// This toolchain exposes the 6-arg builtin form (g0 x4, g1 x8, g2 x4, g3 x4,
// extra x8, cpol).
__device__ __forceinline__ void tdm_load_xg_tile(const float* gptr,
                                                 unsigned lds_byte) {
  unsigned long long ga = (unsigned long long)(size_t)gptr;
  u32x4 g0;
  g0[0] = 1u;                                   // count=1, user load descriptor
  g0[1] = lds_byte;                             // lds_addr (bytes)
  g0[2] = (unsigned)(ga & 0xffffffffu);         // global_addr[31:0]
  g0[3] = (unsigned)((ga >> 32) & 0x1ffffffu)   // global_addr[56:32]
          | (2u << 30);                         // type = 2 ("image")
  i32x8 g1;
  g1[0] = (int)(2u << 16);       // workgroup_mask=0, data_size=2 (4 bytes)
  g1[1] = (int)(1024u << 16);    // tensor_dim0[15:0]=1024 (bits 63:48)
  g1[2] = (int)(4u << 16);       // tensor_dim0[31:16]=0 | tensor_dim1[15:0]=4
  g1[3] = (int)(1024u << 16);    // tensor_dim1[31:16]=0 | tile_dim0=1024
  g1[4] = 4;                     // tile_dim1=4, tile_dim2=0
  g1[5] = 1 << 20;               // tensor_dim0_stride = 1048576 elements (S*GD)
  g1[6] = 0;                     // stride0[47:32]=0, tensor_dim1_stride lo=0
  g1[7] = 0;
  i32x4 gz4 = {};
  i32x8 gz8 = {};
  __builtin_amdgcn_tensor_load_to_lds(g0, g1, gz4, gz4, gz8, 0);
}

// ---------------- elementwise helpers ----------------
__global__ void cvt_f32_to_f16(const float* __restrict__ in,
                               _Float16* __restrict__ out, int n) {
  for (int i = blockIdx.x * blockDim.x + threadIdx.x; i < n;
       i += gridDim.x * blockDim.x)
    out[i] = (_Float16)in[i];
}

__global__ void bias_sum_kernel(const float* __restrict__ a,
                                const float* __restrict__ b,
                                float* __restrict__ o, int n) {
  int i = blockIdx.x * blockDim.x + threadIdx.x;
  if (i < n) o[i] = a[i] + b[i];
}

// ---------------- xg = X @ Wih^T + (bih + bhh) ----------------
__global__ void xg_gemm_kernel(const _Float16* __restrict__ Xb,
                               const _Float16* __restrict__ Wb,
                               const float* __restrict__ Bb2,
                               float* __restrict__ XGb,
                               size_t sX, size_t sW, size_t sB, size_t sXG) {
  const int s = blockIdx.y;
  const _Float16* X = Xb + (size_t)s * sX;
  const _Float16* W = Wb + (size_t)s * sW;
  const float* bias = Bb2 + (size_t)s * sB;
  float* XG = XGb + (size_t)s * sXG;

  const int lane = threadIdx.x & 31, wave = threadIdx.x >> 5;
  const int grp = lane >> 4, nl = lane & 15;
  const int id = blockIdx.x * 8 + wave;           // 16384 tiles total
  const int m0 = (id >> 6) * 16, n0 = (id & 63) * 16;

  const _Float16* arow = X + (size_t)(m0 + nl) * DM;
  const _Float16* brow = W + (size_t)(n0 + nl) * DM;
  __builtin_prefetch(arow, 0, 0);                 // global_prefetch_b8
  __builtin_prefetch(brow, 0, 0);

  v8f acc = {};
#pragma unroll
  for (int kk = 0; kk < 8; ++kk)
    acc = wmma16(load_a_frag(arow, kk * 32, grp),
                 load_b_frag(brow, kk * 32, grp), acc);

  const int n = n0 + nl;
  const float bn = bias[n];
#pragma unroll
  for (int r = 0; r < 8; ++r) {
    const int m = m0 + r + 8 * grp;
    XG[(size_t)m * GD + n] = acc[r] + bn;
  }
}

// ---------------- sequential LSTM scan ----------------
// One block per stream. h lives in LDS as a 16x256 f16 tile (rows 0-3 = batch).
// xg_t tiles (4x1024 f32) are double-buffered into LDS by the TDM one step
// ahead, overlapping the DMA with the recurrent WMMA work.
__global__ void lstm_scan_kernel(const float* __restrict__ xg0,
                                 const _Float16* __restrict__ whh0,
                                 float* __restrict__ h32_0,
                                 _Float16* __restrict__ h16_0,
                                 size_t sXG, size_t sW, size_t sH) {
  const int s = blockIdx.x;
  const float* xg = xg0 + (size_t)s * sXG;
  const _Float16* whh = whh0 + (size_t)s * sW;
  float* h32 = h32_0 + (size_t)s * sH;
  _Float16* h16 = h16_0 + (size_t)s * sH;

  __shared__ __align__(16) _Float16 hsh[16][DM];   // 8 KB
  __shared__ float gsh[BB][GD];                    // 16 KB
  __shared__ float csh[BB][DM];                    // 4 KB
  __shared__ __align__(16) float xgs[2][BB][GD];   // 32 KB TDM double buffer

  const int tid = threadIdx.x;
  for (int i = tid; i < 16 * DM; i += 256) ((_Float16*)hsh)[i] = (_Float16)0.0f;
  for (int i = tid; i < BB * DM; i += 256) ((float*)csh)[i] = 0.0f;

  const int lane = tid & 31, wave = tid >> 5, grp = lane >> 4, nl = lane & 15;

  // prologue: kick off TDM for step 0
  if (wave == 0) tdm_load_xg_tile(xg, (unsigned)(size_t)&xgs[0][0][0]);
  __syncthreads();

  for (int t = 0; t < SQ; ++t) {
    const int cur = t & 1, nxt = cur ^ 1;
    if (wave == 0) __builtin_amdgcn_s_wait_tensorcnt(0);
    __syncthreads();  // xgs[cur] ready for everyone; xgs[nxt] reads all retired
    if (wave == 0 && t + 1 < SQ)
      tdm_load_xg_tile(xg + (size_t)(t + 1) * GD,
                       (unsigned)(size_t)&xgs[nxt][0][0]);

    // A fragments of h (shared by all this wave's N-tiles), read via ds_load
    v16h afrag[8];
    const _Float16* arow = &hsh[nl][0];
#pragma unroll
    for (int kk = 0; kk < 8; ++kk) afrag[kk] = load_a_frag(arow, kk * 32, grp);

#pragma unroll
    for (int nt = 0; nt < 8; ++nt) {
      const int n0 = wave * 128 + nt * 16;
      const _Float16* brow = whh + (size_t)(n0 + nl) * DM;  // L2-resident
      v8f acc = {};
#pragma unroll
      for (int kk = 0; kk < 8; ++kk)
        acc = wmma16(afrag[kk], load_b_frag(brow, kk * 32, grp), acc);
      const int n = n0 + nl;
      if (grp == 0) {  // rows m = r (+8*grp); only m<4 are real batches
#pragma unroll
        for (int r = 0; r < 4; ++r)
          gsh[r][n] = acc[r] + xgs[cur][r][n];
      }
    }
    __syncthreads();

    // cell update: thread tid owns hidden unit j = tid for all 4 batches
#pragma unroll
    for (int b = 0; b < BB; ++b) {
      float gi = gsh[b][tid];
      float gf = gsh[b][tid + 256];
      float gg = gsh[b][tid + 512];
      float go = gsh[b][tid + 768];
      float c = csh[b][tid];
      c = sigf(gf) * c + sigf(gi) * tanhf(gg);
      float h = sigf(go) * tanhf(c);
      csh[b][tid] = c;
      hsh[b][tid] = (_Float16)h;
      const size_t off = ((size_t)b * SQ + t) * DM + tid;
      h32[off] = h;
      h16[off] = (_Float16)h;
    }
    __syncthreads();
  }
}

// ---------------- attention: scores -> exact GELU -> context ----------------
__global__ void attn_kernel(const _Float16* __restrict__ q,
                            const _Float16* __restrict__ k,
                            const _Float16* __restrict__ v,
                            float* __restrict__ ctx32,
                            _Float16* __restrict__ ctx16) {
  __shared__ __align__(16) _Float16 psh[8][16][32];  // probs stage, 8 KB
  __shared__ __align__(16) _Float16 vsh[8][32][32];  // v^T stage, 16 KB
  const int lane = threadIdx.x & 31, wave = threadIdx.x >> 5;
  const int grp = lane >> 4, nl = lane & 15;
  const int id = blockIdx.x * 8 + wave;  // 2048 = 4b * 8h * 64 q-tiles
  const int b = id >> 9, hh = (id >> 6) & 7, qt = id & 63;
  const int m0 = qt * 16, ho = hh * 32;

  _Float16(*probs)[32] = psh[wave];
  _Float16(*vT)[32] = vsh[wave];

  const _Float16* qrow = q + ((size_t)b * SQ + m0 + nl) * DM + ho;
  const v16h aq = load_a_frag(qrow, 0, grp);

  v8f ctx0 = {}, ctx1 = {};
  for (int kc = 0; kc < 32; ++kc) {  // 32 key-chunks of 32
    const int key0 = kc * 32;
    // stage v^T[dk][key] so PV B-fragments are contiguous ds reads
    const _Float16* vrow = v + ((size_t)b * SQ + key0 + lane) * DM + ho;
#pragma unroll
    for (int dk = 0; dk < 32; ++dk) vT[dk][lane] = vrow[dk];

#pragma unroll
    for (int st = 0; st < 2; ++st) {  // two 16-key score tiles
      const _Float16* krow =
          k + ((size_t)b * SQ + key0 + st * 16 + nl) * DM + ho;
      v8f sc = {};
      sc = wmma16(aq, load_b_frag(krow, 0, grp), sc);
#pragma unroll
      for (int r = 0; r < 8; ++r) {
        float x = sc[r] * 0.1767766953f;                      // 1/sqrt(32)
        float p = 0.5f * x * (1.0f + erff(x * 0.70710678f));  // exact GELU
        probs[r + 8 * grp][st * 16 + nl] = (_Float16)p;
      }
    }
    __builtin_amdgcn_wave_barrier();  // order LDS stage vs fragment reads
    const v16h ap = load_a_frag(&probs[nl][0], 0, grp);
    ctx0 = wmma16(ap, load_b_frag(&vT[nl][0], 0, grp), ctx0);
    ctx1 = wmma16(ap, load_b_frag(&vT[16 + nl][0], 0, grp), ctx1);
    __builtin_amdgcn_wave_barrier();
  }
#pragma unroll
  for (int r = 0; r < 8; ++r) {
    const int m = m0 + r + 8 * grp;
    const size_t base = ((size_t)b * SQ + m) * DM + ho;  // combined-head layout
    ctx32[base + nl] = ctx0[r];
    ctx16[base + nl] = (_Float16)ctx0[r];
    ctx32[base + 16 + nl] = ctx1[r];
    ctx16[base + 16 + nl] = (_Float16)ctx1[r];
  }
}

extern "C" void kernel_launch(void* const* d_in, const int* in_sizes, int n_in,
                              void* d_out, int out_size, void* d_ws,
                              size_t ws_size, hipStream_t stream) {
  (void)in_sizes; (void)n_in; (void)out_size; (void)ws_size;
  const float* X[3] = {(const float*)d_in[0], (const float*)d_in[1],
                       (const float*)d_in[2]};
  // d_in[3] = use_taa (fixed 1). Params flatten: per set {Wih, Whh, bih, bhh},
  // sets in order q, k, v, o.
  const float *Wih[4], *Whh[4], *bih[4], *bhh[4];
  for (int s = 0; s < 4; ++s) {
    Wih[s] = (const float*)d_in[4 + 4 * s + 0];
    Whh[s] = (const float*)d_in[4 + 4 * s + 1];
    bih[s] = (const float*)d_in[4 + 4 * s + 2];
    bhh[s] = (const float*)d_in[4 + 4 * s + 3];
  }

  const size_t WELEM = (size_t)GD * DM;        // 262144 per weight matrix
  const size_t XELEM = (size_t)BB * SQ * DM;   // 1048576
  const size_t XGELEM = (size_t)BB * SQ * GD;  // 4194304

  char* p = (char*)d_ws;
  auto take = [&](size_t bytes) -> char* {
    char* r = p;
    p += (bytes + 255) & ~(size_t)255;
    return r;
  };
  // f16 weights: [WihQ, WhhQ, WihK, WhhK, WihV, WhhV, WihO, WhhO]
  _Float16* wf16 = (_Float16*)take(8 * WELEM * sizeof(_Float16));
  float* bias = (float*)take(4 * GD * sizeof(float));
  _Float16* xf16 = (_Float16*)take(3 * XELEM * sizeof(_Float16));
  float* xg = (float*)take(4 * XGELEM * sizeof(float));
  float* h32 = (float*)take(3 * XELEM * sizeof(float));
  _Float16* h16 = (_Float16*)take(3 * XELEM * sizeof(_Float16));
  float* ctx32 = (float*)take(XELEM * sizeof(float));
  _Float16* ctx16 = (_Float16*)take(XELEM * sizeof(_Float16));
  _Float16* h16o = (_Float16*)take(XELEM * sizeof(_Float16));

  for (int s = 0; s < 4; ++s) {
    cvt_f32_to_f16<<<256, 256, 0, stream>>>(Wih[s], wf16 + (2 * s) * WELEM,
                                            (int)WELEM);
    cvt_f32_to_f16<<<256, 256, 0, stream>>>(Whh[s], wf16 + (2 * s + 1) * WELEM,
                                            (int)WELEM);
    bias_sum_kernel<<<4, 256, 0, stream>>>(bih[s], bhh[s], bias + s * GD, GD);
  }
  for (int j = 0; j < 3; ++j)
    cvt_f32_to_f16<<<512, 256, 0, stream>>>(X[j], xf16 + j * XELEM, (int)XELEM);

  // xg for q/k/v streams (three GEMMs in one launch, blockIdx.y = stream)
  xg_gemm_kernel<<<dim3(2048, 3), 256, 0, stream>>>(
      xf16, wf16, bias, xg, XELEM, 2 * WELEM, (size_t)GD, XGELEM);

  // three concurrent recurrent scans (q, k, v)
  lstm_scan_kernel<<<3, 256, 0, stream>>>(xg, wf16 + WELEM, h32, h16, XGELEM,
                                          2 * WELEM, XELEM);

  attn_kernel<<<256, 256, 0, stream>>>(h16, h16 + XELEM, h16 + 2 * XELEM,
                                       ctx32, ctx16);

  // xg for the output LSTM from ctx
  xg_gemm_kernel<<<dim3(2048, 1), 256, 0, stream>>>(
      ctx16, wf16 + 6 * WELEM, bias + 3 * GD, xg + 3 * XGELEM, 0, 0, 0, 0);

  // final scan writes f32 hidden states straight to d_out
  lstm_scan_kernel<<<1, 256, 0, stream>>>(xg + 3 * XGELEM, wf16 + 7 * WELEM,
                                          (float*)d_out, h16o, 0, 0, 0);
}
